// DCRNNCell_65317862638001
// MI455X (gfx1250) — compile-verified, matrix operand order checked by reading, not run on hardware
//
#include <hip/hip_runtime.h>
#include <hip/hip_bf16.h>

// ---------------------------------------------------------------------------
// DCRNN cell for MI455X (gfx1250, wave32).
// Dominant cost: 4x GEMM  Y[4096 x 1056] = S^T[4096x4096] * X[4096x1056]
// -> bf16 WMMA (v_wmma_f32_16x16x32_bf16), f32 accumulate, async-LDS double
//    buffering, one barrier per k-step, 4 WMMA per wave per k-step.
//    Loop body: (1) issue prefetch (async A + B->regs), (2) frag loads +
//    4 WMMAs, (3) B regs -> LDS (loadcnt wait lands after WMMA issue).
// ---------------------------------------------------------------------------

#define NN    4096          // graph nodes (M and K of big GEMM)
#define BB    16            // batch
#define CH    66            // C_in + H = 2 + 64
#define JJ    (BB * CH)     // 1056 valid columns
#define JP    1088          // padded to 64*17 so BN=64 tiles divide evenly
#define HH    64
#define FEAT  198           // (K+1)*CH

typedef __attribute__((ext_vector_type(16))) __bf16 v16bf;
typedef __attribute__((ext_vector_type(8)))  float  v8f;
typedef __attribute__((ext_vector_type(4)))  unsigned u32x4;
typedef __attribute__((ext_vector_type(4)))  int      i32x4;

#define AS1 __attribute__((address_space(1)))
#define AS3 __attribute__((address_space(3)))

// -------- async global->LDS (CDNA5), guarded; fallback = load + ds_store ----
#ifndef USE_ASYNC_LDS
#if defined(__has_builtin)
#if __has_builtin(__builtin_amdgcn_global_load_async_to_lds_b128)
#define USE_ASYNC_LDS 1
#endif
#endif
#endif
#ifndef USE_ASYNC_LDS
#define USE_ASYNC_LDS 0
#endif

__device__ __forceinline__ void copy16_g2l(const __bf16* g, __bf16* l) {
#if USE_ASYNC_LDS
  AS1 i32x4* gp = (AS1 i32x4*)(unsigned long long)(const void*)g;
  AS3 i32x4* lp = (AS3 i32x4*)(unsigned)(unsigned long long)(void*)l;
  __builtin_amdgcn_global_load_async_to_lds_b128(gp, lp, 0, 0);
#else
  *(u32x4*)l = *(const u32x4*)g;
#endif
}

__device__ __forceinline__ void wait_async_lds() {
#if USE_ASYNC_LDS
#if __has_builtin(__builtin_amdgcn_s_wait_asynccnt)
  __builtin_amdgcn_s_wait_asynccnt(0);
#else
  asm volatile("s_wait_asynccnt 0" ::: "memory");
#endif
#endif
}

// ---------------------------------------------------------------------------
// K0: ST[m][n] = (bf16) S[n][m]   (LDS-tiled transpose + f32->bf16 convert)
// ---------------------------------------------------------------------------
__global__ __launch_bounds__(256) void transpose_cvt(const float* __restrict__ S,
                                                     __bf16* __restrict__ ST) {
  __shared__ float tile[32][33];
  const int tx = threadIdx.x & 31, ty = threadIdx.x >> 5;  // ty 0..7
  const int m0 = blockIdx.x * 32, n0 = blockIdx.y * 32;
#pragma unroll
  for (int i = 0; i < 4; ++i) {
    int r = ty + i * 8;
    tile[r][tx] = S[(size_t)(n0 + r) * NN + m0 + tx];
  }
  __syncthreads();
#pragma unroll
  for (int i = 0; i < 4; ++i) {
    int r = ty + i * 8;
    ST[(size_t)(m0 + r) * NN + n0 + tx] = (__bf16)tile[tx][r];
  }
}

// ---------------------------------------------------------------------------
// K1: X0[n][b*66+c] = c<2 ? x[b,n,c] : h[b,n,c-2]; pad cols [1056,1088) = 0
// ---------------------------------------------------------------------------
__global__ __launch_bounds__(256) void pack_x0(const float* __restrict__ x,
                                               const float* __restrict__ h,
                                               __bf16* __restrict__ X0) {
  size_t idx = (size_t)blockIdx.x * 256 + threadIdx.x;
  if (idx >= (size_t)NN * JP) return;
  int n = (int)(idx / JP);
  int j = (int)(idx % JP);
  float v = 0.f;
  if (j < JJ) {
    int b = j / CH, c = j % CH;
    v = (c < 2) ? x[((size_t)b * NN + n) * 2 + c]
                : h[((size_t)b * NN + n) * HH + (c - 2)];
  }
  X0[idx] = (__bf16)v;
}

// ---------------------------------------------------------------------------
// K2: Out = alpha * (A * B) + beta * Csub     (all bf16, f32 accumulate)
//   A : [4096][4096] row-major (= S^T);  B : [4096][JP]
//   Tile: BM=128, BN=64, BK=32; 256 threads (8 waves).
// ---------------------------------------------------------------------------
#define BM 128
#define BN 64
#define BK 32
#define LDA 40   // padded LDS row stride (bf16 elems): 80B, 16B aligned

union Frag32B { struct { u32x4 a, b; } q; v16bf v; };

#define WMMA_BF16(A_, B_, C_)                                            \
  __builtin_amdgcn_wmma_f32_16x16x32_bf16(false, (A_), false, (B_),      \
                                          (short)0, (C_), false, false)

__global__ __launch_bounds__(256) void gemm_bf16(const __bf16* __restrict__ A,
                                                 const __bf16* __restrict__ Bm,
                                                 const __bf16* __restrict__ Csub,
                                                 __bf16* __restrict__ Out,
                                                 float alpha, float beta) {
  __shared__ __align__(16) __bf16 lA[2 * BM * LDA];  // 2 x 10 KB
  __shared__ __align__(16) __bf16 lB[2 * BN * LDA];  // 2 x  5 KB

  const int t = threadIdx.x;
  const int lane = t & 31, wv = t >> 5;        // 8 waves
  const int lm = lane & 15, kh = lane >> 4;    // WMMA per-lane addressing
  const int mBase = blockIdx.x * BM;
  const int jBase = blockIdx.y * BN;
  const int kl = t >> 3;                       // B staging: k row 0..31
  const int jg = (t & 7) * 8;                  // B staging: j group 0..56

  v8f acc0 = {}, acc1 = {}, acc2 = {}, acc3 = {};

  // ---- prologue: stage tile 0 completely -------------------------------
  {
#pragma unroll
    for (int i = 0; i < 2; ++i) {
      int cidx = t + i * 256;
      int row = cidx >> 2;
      int off = (cidx & 3) * 8;
      copy16_g2l(A + (size_t)(mBase + row) * NN + off, lA + row * LDA + off);
    }
    union { u32x4 u; __bf16 e[8]; } q;
    q.u = *(const u32x4*)(Bm + (size_t)kl * JP + jBase + jg);
#pragma unroll
    for (int c = 0; c < 8; ++c) lB[(jg + c) * LDA + kl] = q.e[c];
  }

  const int NT = NN / BK;  // 128
#pragma unroll 1           // single body -> fixed acc registers, no rotation
  for (int kt = 0; kt < NT; ++kt) {
    const int cur = kt & 1;
    const int nxt = cur ^ 1;
    wait_async_lds();     // tile kt resident (only its asyncs outstanding)
    __syncthreads();      // also orders late lB ds_stores of prev iteration

    const bool pf = (kt + 1 < NT);
    union { u32x4 u; __bf16 e[8]; } q;

    // (1) early prefetch issue: async A copies + B tile -> registers
    if (pf) {
      const int k0n = (kt + 1) * BK;
      __bf16* la = lA + nxt * (BM * LDA);
#pragma unroll
      for (int i = 0; i < 2; ++i) {
        int cidx = t + i * 256;
        int row = cidx >> 2;
        int off = (cidx & 3) * 8;
        copy16_g2l(A + (size_t)(mBase + row) * NN + k0n + off,
                   la + row * LDA + off);
      }
      q.u = *(const u32x4*)(Bm + (size_t)(k0n + kl) * JP + jBase + jg);
    }

    // (2) compute on cur: preload A + 4 B fragments, then 4 WMMAs
    const __bf16* la = lA + cur * (BM * LDA);
    const __bf16* lb = lB + cur * (BN * LDA);
    Frag32B fa, fb0, fb1, fb2, fb3;
    const __bf16* ap = la + (wv * 16 + lm) * LDA + kh * 8;
    fa.q.a = *(const u32x4*)(ap);
    fa.q.b = *(const u32x4*)(ap + 16);
    const __bf16* bp = lb + lm * LDA + kh * 8;  // f*16*LDA are imm offsets
    fb0.q.a = *(const u32x4*)(bp);
    fb0.q.b = *(const u32x4*)(bp + 16);
    fb1.q.a = *(const u32x4*)(bp + 16 * LDA);
    fb1.q.b = *(const u32x4*)(bp + 16 * LDA + 16);
    fb2.q.a = *(const u32x4*)(bp + 32 * LDA);
    fb2.q.b = *(const u32x4*)(bp + 32 * LDA + 16);
    fb3.q.a = *(const u32x4*)(bp + 48 * LDA);
    fb3.q.b = *(const u32x4*)(bp + 48 * LDA + 16);
    acc0 = WMMA_BF16(fa.v, fb0.v, acc0);
    acc1 = WMMA_BF16(fa.v, fb1.v, acc1);
    acc2 = WMMA_BF16(fa.v, fb2.v, acc2);
    acc3 = WMMA_BF16(fa.v, fb3.v, acc3);

    // (3) late B scatter to LDS: s_wait_loadcnt lands after WMMA issue,
    //     giving the global load the whole compute phase to complete.
    if (pf) {
      __bf16* lbn = lB + nxt * (BN * LDA);
#pragma unroll
      for (int c = 0; c < 8; ++c) lbn[(jg + c) * LDA + kl] = q.e[c];
    }
  }

  // epilogue: C/D layout -> element (M = r + 8*(lane>=16), N = lane&15)
  const int mrow = mBase + wv * 16 + kh * 8;
  v8f accs[4] = {acc0, acc1, acc2, acc3};
#pragma unroll
  for (int f = 0; f < 4; ++f) {
    int j = jBase + f * 16 + lm;
#pragma unroll
    for (int r = 0; r < 8; ++r) {
      size_t o = (size_t)(mrow + r) * JP + j;
      float v = alpha * accs[f][r];
      if (Csub) v += beta * (float)Csub[o];
      Out[o] = (__bf16)v;
    }
  }
}

// ---------------------------------------------------------------------------
// K3: gates = sigmoid(diff @ gate_w.T + gate_b); write u; repack concat(x, r*h)
//   Block: 256 threads, 32 consecutive (b,n) rows (same b). Weights bf16 in LDS.
// ---------------------------------------------------------------------------
__global__ __launch_bounds__(256) void gates_kernel(
    const __bf16* __restrict__ X0, const __bf16* __restrict__ X1,
    const __bf16* __restrict__ X2, const float* __restrict__ gw,
    const float* __restrict__ gb, const float* __restrict__ x,
    const float* __restrict__ h, __bf16* __restrict__ XR0,
    float* __restrict__ U) {
  __shared__ __bf16 wL[128 * FEAT];   // 49.5 KB
  __shared__ __bf16 fL[32][FEAT];     // 12.4 KB
  const int t = threadIdx.x;
  const int row0 = blockIdx.x * 32;
  const int b = row0 >> 12;

  for (int i = t; i < 128 * FEAT; i += 256) wL[i] = (__bf16)gw[i];
  for (int i = t; i < 32 * FEAT; i += 256) {
    int r = i / FEAT, k = i % FEAT;
    int n = (row0 + r) & (NN - 1);
    int s = k / CH, c = k % CH;
    const __bf16* Xs = (s == 0) ? X0 : (s == 1) ? X1 : X2;
    fL[r][k] = Xs[(size_t)n * JP + b * CH + c];
  }
  __syncthreads();

  const int o = t & 127;
  const int g = t >> 7;  // 0..1 -> 16 rows each
  float acc[16];
  float bias = gb[o];
#pragma unroll
  for (int rr = 0; rr < 16; ++rr) acc[rr] = bias;
  for (int k = 0; k < FEAT; ++k) {
    float w = (float)wL[o * FEAT + k];
#pragma unroll
    for (int rr = 0; rr < 16; ++rr) acc[rr] += w * (float)fL[g * 16 + rr][k];
  }
#pragma unroll
  for (int rr = 0; rr < 16; ++rr) {
    int row = row0 + g * 16 + rr;
    int n = row & (NN - 1);
    float sg = 1.f / (1.f + __expf(-acc[rr]));
    if (o < HH) {  // r-gate -> repack concat(x, r*h)
      float rh = sg * h[(size_t)row * HH + o];
      XR0[(size_t)n * JP + b * CH + 2 + o] = (__bf16)rh;
      if (o < 2)
        XR0[(size_t)n * JP + b * CH + o] = (__bf16)x[(size_t)row * 2 + o];
    } else {       // u-gate
      U[(size_t)row * HH + (o - HH)] = sg;
    }
  }
}

// ---------------------------------------------------------------------------
// K4: c = tanh(diff_r @ cand_w.T + cand_b); new_h = u*h + (1-u)*c
// ---------------------------------------------------------------------------
__global__ __launch_bounds__(256) void cand_kernel(
    const __bf16* __restrict__ X0, const __bf16* __restrict__ X1,
    const __bf16* __restrict__ X2, const float* __restrict__ cw,
    const float* __restrict__ cb, const float* __restrict__ h,
    const float* __restrict__ U, float* __restrict__ out) {
  __shared__ __bf16 wL[HH * FEAT];    // 24.8 KB
  __shared__ __bf16 fL[32][FEAT];     // 12.4 KB
  const int t = threadIdx.x;
  const int row0 = blockIdx.x * 32;
  const int b = row0 >> 12;

  for (int i = t; i < HH * FEAT; i += 256) wL[i] = (__bf16)cw[i];
  for (int i = t; i < 32 * FEAT; i += 256) {
    int r = i / FEAT, k = i % FEAT;
    int n = (row0 + r) & (NN - 1);
    int s = k / CH, c = k % CH;
    const __bf16* Xs = (s == 0) ? X0 : (s == 1) ? X1 : X2;
    fL[r][k] = Xs[(size_t)n * JP + b * CH + c];
  }
  __syncthreads();

  const int o = t & 63;
  const int g = t >> 6;  // 0..3 -> 8 rows each
  float acc[8];
  float bias = cb[o];
#pragma unroll
  for (int rr = 0; rr < 8; ++rr) acc[rr] = bias;
  for (int k = 0; k < FEAT; ++k) {
    float w = (float)wL[o * FEAT + k];
#pragma unroll
    for (int rr = 0; rr < 8; ++rr) acc[rr] += w * (float)fL[g * 8 + rr][k];
  }
#pragma unroll
  for (int rr = 0; rr < 8; ++rr) {
    int row = row0 + g * 8 + rr;
    float e2 = __expf(2.f * acc[rr]);
    float c = (e2 - 1.f) / (e2 + 1.f);  // tanh
    float u = U[(size_t)row * HH + o];
    out[(size_t)row * HH + o] = u * h[(size_t)row * HH + o] + (1.f - u) * c;
  }
}

// ---------------------------------------------------------------------------
extern "C" void kernel_launch(void* const* d_in, const int* in_sizes, int n_in,
                              void* d_out, int out_size, void* d_ws,
                              size_t ws_size, hipStream_t stream) {
  const float* x  = (const float*)d_in[0];
  const float* h  = (const float*)d_in[1];
  const float* S  = (const float*)d_in[2];
  const float* gw = (const float*)d_in[3];
  const float* gb = (const float*)d_in[4];
  const float* cw = (const float*)d_in[5];
  const float* cb = (const float*)d_in[6];
  float* out = (float*)d_out;

  // workspace layout (~77 MB): ST | P0 | P1 | P2 | U
  char* ws = (char*)d_ws;
  const size_t szST = (size_t)NN * NN * 2;
  const size_t szP  = (size_t)NN * JP * 2;
  __bf16* ST = (__bf16*)ws;                 ws += szST;
  __bf16* P0 = (__bf16*)ws;                 ws += szP;
  __bf16* P1 = (__bf16*)ws;                 ws += szP;
  __bf16* P2 = (__bf16*)ws;                 ws += szP;
  float*  U  = (float*)ws;

  // S^T in bf16, pack X0 = concat(x,h)
  transpose_cvt<<<dim3(NN / 32, NN / 32), 256, 0, stream>>>(S, ST);
  pack_x0<<<(unsigned)(((size_t)NN * JP + 255) / 256), 256, 0, stream>>>(x, h, P0);

  const dim3 gg(NN / BM, JP / BN);  // 32 x 17
  // diffusion 1: x1 = S^T x0 ; x2 = 2 S^T x1 - x0
  gemm_bf16<<<gg, 256, 0, stream>>>(ST, P0, nullptr, P1, 1.f, 0.f);
  gemm_bf16<<<gg, 256, 0, stream>>>(ST, P1, P0, P2, 2.f, -1.f);
  // gates + repack concat(x, r*h) into P0 (block-local rows -> safe alias)
  gates_kernel<<<(BB * NN) / 32, 256, 0, stream>>>(P0, P1, P2, gw, gb, x, h, P0, U);
  // diffusion 2
  gemm_bf16<<<gg, 256, 0, stream>>>(ST, P0, nullptr, P1, 1.f, 0.f);
  gemm_bf16<<<gg, 256, 0, stream>>>(ST, P1, P0, P2, 2.f, -1.f);
  // candidate + GRU combine
  cand_kernel<<<(BB * NN) / 32, 256, 0, stream>>>(P0, P1, P2, cw, cb, h, U, out);
}
